// TransformerDecoderBlock_22127671509219
// MI455X (gfx1250) — compile-verified, matrix operand order checked by reading, not run on hardware
//
#include <hip/hip_runtime.h>
#include <hip/hip_bf16.h>

#define DEV __device__ __forceinline__

typedef __attribute__((ext_vector_type(16))) _Float16 v16h;
typedef __attribute__((ext_vector_type(8)))  _Float16 v8h;
typedef __attribute__((ext_vector_type(4)))  _Float16 v4h;
typedef __attribute__((ext_vector_type(8)))  float    v8f;
typedef __attribute__((ext_vector_type(4)))  float    v4f;

static constexpr int NB   = 4;
static constexpr int TT   = 2048;   // T == S
static constexpr int EE   = 512;
static constexpr int HH   = 8;
static constexpr int DH   = 64;
static constexpr int DFF  = 2048;
static constexpr int MR   = NB * TT; // 8192 rows

// D = A(16x32 f16) * B(32x16 f16) + C(16x16 f32)
#define WMMA(A, B, C) \
  __builtin_amdgcn_wmma_f32_16x16x32_f16(false, (A), false, (B), (short)0, (C), false, false)

DEV v8f zero8() {
  v8f z;
#pragma unroll
  for (int i = 0; i < 8; ++i) z[i] = 0.0f;
  return z;
}

// Load a 16x32 (rows x K) f16 fragment from row-major memory with leading dim ld.
// Per ISA layout: lanes 0-15 hold row=lane, K {0..7, 16..23}; lanes 16-31 hold
// row=lane-16, K {8..15, 24..31}.  Two b128 loads per lane.
DEV v16h load_frag(const _Float16* __restrict__ base, int ld) {
  const int lane = threadIdx.x & 31;
  const int r = lane & 15;
  const int sel = lane >> 4;
  const _Float16* p = base + (size_t)r * ld + sel * 8;
  v8h lo = *(const v8h*)p;
  v8h hi = *(const v8h*)(p + 16);
  v16h f;
#pragma unroll
  for (int i = 0; i < 8; ++i) { f[i] = lo[i]; f[i + 8] = hi[i]; }
  return f;
}

// ---------------------------------------------------------------------------
// f32 -> f16 conversion (weights and activations), vectorized x4
// ---------------------------------------------------------------------------
__global__ void f32_to_f16(const float* __restrict__ in, _Float16* __restrict__ out, int n4) {
  int i = blockIdx.x * blockDim.x + threadIdx.x;
  if (i < n4) {
    v4f v = ((const v4f*)in)[i];
    v4h o;
#pragma unroll
    for (int k = 0; k < 4; ++k) o[k] = (_Float16)v[k];
    ((v4h*)out)[i] = o;
  }
}

// ---------------------------------------------------------------------------
// NT GEMM: out[m,n] = sum_k A[m,k]*B[n,k] + bias[n]  (A: MxK f16, B: NnxK f16)
// Block = 256 threads = 8 waves, block tile 64(M) x 128(N), wave tile 32x32.
// Epilogue modes select output layout / fusion.
// ---------------------------------------------------------------------------
enum { EP_HEADED_QK = 0, EP_HEADED_VT = 1, EP_F32_RESID = 2, EP_F16_RELU = 3 };

template <int MODE>
__global__ __launch_bounds__(256) void gemm_nt(
    const _Float16* __restrict__ A, const _Float16* __restrict__ B,
    const float* __restrict__ bias, const float* __restrict__ resid,
    void* __restrict__ out, int Nn, int K, float qscale) {
  const int w    = threadIdx.x >> 5;
  const int lane = threadIdx.x & 31;
  const int m0 = blockIdx.y * 64 + (w >> 2) * 32;
  const int n0 = blockIdx.x * 128 + (w & 3) * 32;

  v8f c00 = zero8(), c01 = zero8(), c10 = zero8(), c11 = zero8();
  for (int k = 0; k < K; k += 32) {
    v16h a0 = load_frag(A + (size_t)m0 * K + k, K);
    v16h a1 = load_frag(A + (size_t)(m0 + 16) * K + k, K);
    v16h b0 = load_frag(B + (size_t)n0 * K + k, K);
    v16h b1 = load_frag(B + (size_t)(n0 + 16) * K + k, K);
    c00 = WMMA(a0, b0, c00);
    c01 = WMMA(a0, b1, c01);
    c10 = WMMA(a1, b0, c10);
    c11 = WMMA(a1, b1, c11);
  }

  const int nl = lane & 15;          // column within 16x16 tile
  const int mh = (lane >> 4) * 8;    // row-half offset
  v8f acc[2][2] = {{c00, c01}, {c10, c11}};

#pragma unroll
  for (int i = 0; i < 2; ++i) {
#pragma unroll
    for (int j = 0; j < 2; ++j) {
      const int n  = n0 + 16 * j + nl;
      const int mb = m0 + 16 * i + mh;
      const float bn = bias[n];
      if constexpr (MODE == EP_F32_RESID) {
        float* o = (float*)out;
#pragma unroll
        for (int e = 0; e < 8; ++e) {
          const size_t idx = (size_t)(mb + e) * Nn + n;
          o[idx] = acc[i][j][e] + bn + resid[idx];
        }
      } else if constexpr (MODE == EP_F16_RELU) {
        _Float16* o = (_Float16*)out;
#pragma unroll
        for (int e = 0; e < 8; ++e) {
          float v = acc[i][j][e] + bn;
          v = v > 0.0f ? v : 0.0f;
          o[(size_t)(mb + e) * Nn + n] = (_Float16)v;
        }
      } else if constexpr (MODE == EP_HEADED_QK) {
        // out layout (batch, head, t, dh), t-length = TT, with optional scale
        _Float16* o = (_Float16*)out;
        const int h = n >> 6, d = n & 63;
#pragma unroll
        for (int e = 0; e < 8; ++e) {
          const int m = mb + e;
          const int bb = m >> 11, t = m & (TT - 1);
          o[(((size_t)(bb * HH + h)) * TT + t) * DH + d] =
              (_Float16)((acc[i][j][e] + bn) * qscale);
        }
      } else { // EP_HEADED_VT: out layout (batch, head, dh, t) -> contiguous in t
        _Float16* o = (_Float16*)out;
        const int h = n >> 6, d = n & 63;
        const int bb = mb >> 11, t0 = mb & (TT - 1);
        v8h pk;
#pragma unroll
        for (int e = 0; e < 8; ++e) pk[e] = (_Float16)(acc[i][j][e] + bn);
        *(v8h*)&o[(((size_t)(bb * HH + h)) * DH + d) * TT + t0] = pk;
      }
    }
  }
}

// ---------------------------------------------------------------------------
// Flash attention (streaming softmax), one wave per 16-query tile per head.
// Scores computed transposed (S^T = K * Q^T) so row stats are per-lane and the
// exp()'d probabilities land directly in the B-fragment layout for P@V.
// Q is pre-scaled by 1/sqrt(DH); V is pre-transposed (b,h,d,t).
// ---------------------------------------------------------------------------
template <bool CAUSAL>
__global__ __launch_bounds__(256) void flash_attn(
    const _Float16* __restrict__ Q, const _Float16* __restrict__ Kk,
    const _Float16* __restrict__ Vt, _Float16* __restrict__ Oh, int Slen) {
  const int w    = threadIdx.x >> 5;
  const int lane = threadIdx.x & 31;
  const int b = blockIdx.z, h = blockIdx.y;
  const int t0 = (blockIdx.x * 8 + w) * 16;

  const _Float16* Qb = Q + (((size_t)(b * HH + h)) * TT + t0) * DH;
  const _Float16* Kb = Kk + ((size_t)(b * HH + h)) * TT * DH;
  const _Float16* Vb = Vt + ((size_t)(b * HH + h)) * DH * TT;

  const v16h q0 = load_frag(Qb + 0, DH);   // d = 0..31
  const v16h q1 = load_frag(Qb + 32, DH);  // d = 32..63

  v8f o0 = zero8(), o1 = zero8(), o2 = zero8(), o3 = zero8();
  float mrun = -3.0e38f, lrun = 0.0f;

  const int tl = lane & 15;
  const int shalf = (lane >> 4) * 8;
  const int tg = t0 + tl;
  const int send = CAUSAL ? (t0 + 16) : Slen;

  for (int s0 = 0; s0 < send; s0 += 32) {
    // score tiles (transposed): st0 -> s in [s0, s0+16), st1 -> [s0+16, s0+32)
    v8f st0 = zero8(), st1 = zero8();
    {
      v16h k0 = load_frag(Kb + (size_t)s0 * DH, DH);
      v16h k1 = load_frag(Kb + (size_t)s0 * DH + 32, DH);
      st0 = WMMA(k0, q0, st0);
      st0 = WMMA(k1, q1, st0);
      v16h k2 = load_frag(Kb + (size_t)(s0 + 16) * DH, DH);
      v16h k3 = load_frag(Kb + (size_t)(s0 + 16) * DH + 32, DH);
      st1 = WMMA(k2, q0, st1);
      st1 = WMMA(k3, q1, st1);
    }
    if (CAUSAL && (s0 + 31 > t0)) {
#pragma unroll
      for (int e = 0; e < 8; ++e) {
        const int sA = s0 + shalf + e;
        if (sA > tg) st0[e] = -1.0e30f;
        if (sA + 16 > tg) st1[e] = -1.0e30f;
      }
    }
    // streaming softmax over this 32-wide chunk (per-lane + one xor-16 merge)
    float cm = -3.0e38f;
#pragma unroll
    for (int e = 0; e < 8; ++e) cm = fmaxf(cm, fmaxf(st0[e], st1[e]));
    cm = fmaxf(cm, __shfl_xor(cm, 16, 32));
    const float nm = fmaxf(mrun, cm);
    const float alpha = __expf(mrun - nm);
    float csum = 0.0f;
    v16h p;
#pragma unroll
    for (int e = 0; e < 8; ++e) {
      const float p0 = __expf(st0[e] - nm);
      const float p1 = __expf(st1[e] - nm);
      csum += p0 + p1;
      p[e] = (_Float16)p0;
      p[e + 8] = (_Float16)p1;
    }
    csum += __shfl_xor(csum, 16, 32);
    lrun = lrun * alpha + csum;
    mrun = nm;
#pragma unroll
    for (int e = 0; e < 8; ++e) {
      o0[e] *= alpha; o1[e] *= alpha; o2[e] *= alpha; o3[e] *= alpha;
    }
    // O^T(d,t) += V^T(d,s) * P^T(s,t)
    o0 = WMMA(load_frag(Vb + (size_t)0  * TT + s0, TT), p, o0);
    o1 = WMMA(load_frag(Vb + (size_t)16 * TT + s0, TT), p, o1);
    o2 = WMMA(load_frag(Vb + (size_t)32 * TT + s0, TT), p, o2);
    o3 = WMMA(load_frag(Vb + (size_t)48 * TT + s0, TT), p, o3);
  }

  const float inv = 1.0f / lrun;
  _Float16* ob = Oh + ((size_t)(b * TT + tg)) * EE + h * DH + shalf;
  v8h pk;
#pragma unroll
  for (int e = 0; e < 8; ++e) pk[e] = (_Float16)(o0[e] * inv);
  *(v8h*)(ob + 0) = pk;
#pragma unroll
  for (int e = 0; e < 8; ++e) pk[e] = (_Float16)(o1[e] * inv);
  *(v8h*)(ob + 16) = pk;
#pragma unroll
  for (int e = 0; e < 8; ++e) pk[e] = (_Float16)(o2[e] * inv);
  *(v8h*)(ob + 32) = pk;
#pragma unroll
  for (int e = 0; e < 8; ++e) pk[e] = (_Float16)(o3[e] * inv);
  *(v8h*)(ob + 48) = pk;
}

// ---------------------------------------------------------------------------
// LayerNorm over E=512 (one block of 128 threads per row, two-pass variance).
// Writes fp32 output and optionally an f16 copy for the next GEMM.
// ---------------------------------------------------------------------------
__global__ __launch_bounds__(128) void layernorm(
    const float* __restrict__ y, const float* __restrict__ w,
    const float* __restrict__ b, float* __restrict__ xo,
    _Float16* __restrict__ ho) {
  __shared__ float red[4];
  const int row = blockIdx.x;
  const int t = threadIdx.x;
  const float* p = y + (size_t)row * EE;
  v4f v = ((const v4f*)p)[t];

  float s = v[0] + v[1] + v[2] + v[3];
#pragma unroll
  for (int o = 16; o > 0; o >>= 1) s += __shfl_xor(s, o, 32);
  if ((t & 31) == 0) red[t >> 5] = s;
  __syncthreads();
  s = red[0] + red[1] + red[2] + red[3];
  const float mean = s * (1.0f / EE);

  const float d0 = v[0] - mean, d1 = v[1] - mean, d2 = v[2] - mean, d3 = v[3] - mean;
  float q = d0 * d0 + d1 * d1 + d2 * d2 + d3 * d3;
#pragma unroll
  for (int o = 16; o > 0; o >>= 1) q += __shfl_xor(q, o, 32);
  __syncthreads();
  if ((t & 31) == 0) red[t >> 5] = q;
  __syncthreads();
  q = red[0] + red[1] + red[2] + red[3];
  const float inv = rsqrtf(q * (1.0f / EE) + 1e-5f);

  v4f wv = ((const v4f*)w)[t];
  v4f bv = ((const v4f*)b)[t];
  v4f o;
  o[0] = d0 * inv * wv[0] + bv[0];
  o[1] = d1 * inv * wv[1] + bv[1];
  o[2] = d2 * inv * wv[2] + bv[2];
  o[3] = d3 * inv * wv[3] + bv[3];
  ((v4f*)(xo + (size_t)row * EE))[t] = o;
  if (ho) {
    v4h oh;
#pragma unroll
    for (int k = 0; k < 4; ++k) oh[k] = (_Float16)o[k];
    ((v4h*)(ho + (size_t)row * EE))[t] = oh;
  }
}

// ---------------------------------------------------------------------------
// Host-side orchestration
// ---------------------------------------------------------------------------
extern "C" void kernel_launch(void* const* d_in, const int* in_sizes, int n_in,
                              void* d_out, int out_size, void* d_ws, size_t ws_size,
                              hipStream_t stream) {
  (void)in_sizes; (void)n_in; (void)out_size; (void)ws_size;

  const float* tgt   = (const float*)d_in[0];
  const float* mem   = (const float*)d_in[1];
  // d_in[2] = tgt_mask: exact causal tril, implemented analytically in flash_attn
  const float* sa_wq = (const float*)d_in[3];  const float* sa_bq = (const float*)d_in[4];
  const float* sa_wk = (const float*)d_in[5];  const float* sa_bk = (const float*)d_in[6];
  const float* sa_wv = (const float*)d_in[7];  const float* sa_bv = (const float*)d_in[8];
  const float* sa_wo = (const float*)d_in[9];  const float* sa_bo = (const float*)d_in[10];
  const float* ca_wq = (const float*)d_in[11]; const float* ca_bq = (const float*)d_in[12];
  const float* ca_wk = (const float*)d_in[13]; const float* ca_bk = (const float*)d_in[14];
  const float* ca_wv = (const float*)d_in[15]; const float* ca_bv = (const float*)d_in[16];
  const float* ca_wo = (const float*)d_in[17]; const float* ca_bo = (const float*)d_in[18];
  const float* w1    = (const float*)d_in[19]; const float* b1    = (const float*)d_in[20];
  const float* w2    = (const float*)d_in[21]; const float* b2    = (const float*)d_in[22];
  const float* ln1w  = (const float*)d_in[23]; const float* ln1b  = (const float*)d_in[24];
  const float* ln2w  = (const float*)d_in[25]; const float* ln2b  = (const float*)d_in[26];
  const float* ln3w  = (const float*)d_in[27]; const float* ln3b  = (const float*)d_in[28];

  // ---- workspace layout (halves first, then floats); total ~92.3 MB --------
  _Float16* ws = (_Float16*)d_ws;
  const size_t WSZ = (size_t)EE * EE;        // 262144 halves
  const size_t WF  = (size_t)DFF * EE;       // 1048576 halves
  const size_t ACT = (size_t)MR * EE;        // 4194304 halves

  _Float16* h_sa_wq = ws + 0 * WSZ;
  _Float16* h_sa_wk = ws + 1 * WSZ;
  _Float16* h_sa_wv = ws + 2 * WSZ;
  _Float16* h_sa_wo = ws + 3 * WSZ;
  _Float16* h_ca_wq = ws + 4 * WSZ;
  _Float16* h_ca_wk = ws + 5 * WSZ;
  _Float16* h_ca_wv = ws + 6 * WSZ;
  _Float16* h_ca_wo = ws + 7 * WSZ;
  _Float16* h_w1    = ws + 8 * WSZ;
  _Float16* h_w2    = h_w1 + WF;
  _Float16* act_h   = h_w2 + WF;        // tgt_h -> x1_h -> x2_h (f16 GEMM input)
  _Float16* mem_h   = act_h + ACT;
  _Float16* Qh      = mem_h + ACT;      // (b,h,t,d)
  _Float16* Kh      = Qh + ACT;         // (b,h,s,d)
  _Float16* Vt      = Kh + ACT;         // (b,h,d,s)
  _Float16* attn_h  = Vt + ACT;         // (b*t, e)
  _Float16* ffn_h   = Qh;               // aliases Qh..attn_h end: MR*DFF halves exactly
  float*    yb      = (float*)(attn_h + ACT);  // residual sum, fp32
  float*    xb      = yb + ACT;                // LN output, fp32

  const dim3 blk(256);
  const dim3 gE(EE / 128, MR / 64);    // Nn = 512
  const dim3 gF(DFF / 128, MR / 64);   // Nn = 2048
  const dim3 gA(TT / 128, HH, NB);     // flash attention
  const float qs = 0.125f;             // 1/sqrt(DH)

  auto cvt = [&](const float* src, _Float16* dst, size_t n) {
    int n4 = (int)(n / 4);
    f32_to_f16<<<dim3((n4 + 255) / 256), 256, 0, stream>>>(src, dst, n4);
  };

  // ---- precision conversion ----
  cvt(tgt, act_h, ACT);
  cvt(mem, mem_h, ACT);
  cvt(sa_wq, h_sa_wq, WSZ); cvt(sa_wk, h_sa_wk, WSZ);
  cvt(sa_wv, h_sa_wv, WSZ); cvt(sa_wo, h_sa_wo, WSZ);
  cvt(ca_wq, h_ca_wq, WSZ); cvt(ca_wk, h_ca_wk, WSZ);
  cvt(ca_wv, h_ca_wv, WSZ); cvt(ca_wo, h_ca_wo, WSZ);
  cvt(w1, h_w1, WF); cvt(w2, h_w2, WF);

  // ---- stage 1: masked self-attention + residual + LN1 ----
  gemm_nt<EP_HEADED_QK><<<gE, blk, 0, stream>>>(act_h, h_sa_wq, sa_bq, nullptr, Qh, EE, EE, qs);
  gemm_nt<EP_HEADED_QK><<<gE, blk, 0, stream>>>(act_h, h_sa_wk, sa_bk, nullptr, Kh, EE, EE, 1.0f);
  gemm_nt<EP_HEADED_VT><<<gE, blk, 0, stream>>>(act_h, h_sa_wv, sa_bv, nullptr, Vt, EE, EE, 1.0f);
  flash_attn<true><<<gA, blk, 0, stream>>>(Qh, Kh, Vt, attn_h, TT);
  gemm_nt<EP_F32_RESID><<<gE, blk, 0, stream>>>(attn_h, h_sa_wo, sa_bo, tgt, yb, EE, EE, 1.0f);
  layernorm<<<MR, 128, 0, stream>>>(yb, ln1w, ln1b, xb, act_h);

  // ---- stage 2: cross-attention + residual + LN2 ----
  gemm_nt<EP_HEADED_QK><<<gE, blk, 0, stream>>>(act_h, h_ca_wq, ca_bq, nullptr, Qh, EE, EE, qs);
  gemm_nt<EP_HEADED_QK><<<gE, blk, 0, stream>>>(mem_h, h_ca_wk, ca_bk, nullptr, Kh, EE, EE, 1.0f);
  gemm_nt<EP_HEADED_VT><<<gE, blk, 0, stream>>>(mem_h, h_ca_wv, ca_bv, nullptr, Vt, EE, EE, 1.0f);
  flash_attn<false><<<gA, blk, 0, stream>>>(Qh, Kh, Vt, attn_h, TT);
  gemm_nt<EP_F32_RESID><<<gE, blk, 0, stream>>>(attn_h, h_ca_wo, ca_bo, xb, yb, EE, EE, 1.0f);
  layernorm<<<MR, 128, 0, stream>>>(yb, ln2w, ln2b, xb, act_h);

  // ---- stage 3: FFN (ReLU) + residual + LN3 ----
  gemm_nt<EP_F16_RELU><<<gF, blk, 0, stream>>>(act_h, h_w1, b1, nullptr, ffn_h, DFF, EE, 1.0f);
  gemm_nt<EP_F32_RESID><<<gE, blk, 0, stream>>>(ffn_h, h_w2, b2, xb, yb, EE, DFF, 1.0f);
  layernorm<<<MR, 128, 0, stream>>>(yb, ln3w, ln3b, (float*)d_out, nullptr);
}
